// AutoregressiveLSTMDecoder_82497731822119
// MI455X (gfx1250) — matrix operand order repaired and checked by reading, not compiled, not run on hardware
//
#include <hip/hip_runtime.h>
#include <hip/hip_bf16.h>
#include <math.h>

// Problem constants (from reference)
#define B_   32
#define TD_  128
#define TE_  128
#define H_   128
#define D_   128
#define E_   128
#define G_   512    // 4*H
#define VT_  100
#define VP_  100
#define VPP_ 112    // VP padded to 7 tiles of 16

typedef __attribute__((ext_vector_type(16))) __bf16 v16bf;
typedef __attribute__((ext_vector_type(8)))  __bf16 v8bf;
typedef __attribute__((ext_vector_type(8)))  float  v8f;

// ---------------------------------------------------------------------------
// Branchless activations on the hardware transcendental pipe
// (v_exp_f32 / v_rcp_f32) — no exec-mask divergence, ~7 VALU ops each.
// ---------------------------------------------------------------------------
#define LOG2E_ 1.4426950408889634f
__device__ __forceinline__ float fast_sigmoid(float x) {
    float e = __builtin_amdgcn_exp2f(-LOG2E_ * x);
    return __builtin_amdgcn_rcpf(1.0f + e);
}
__device__ __forceinline__ float fast_tanh(float x) {
    float e = __builtin_amdgcn_exp2f(2.8853900817779268f * x);   // exp(2x)
    return 1.0f - 2.0f * __builtin_amdgcn_rcpf(e + 1.0f);
}
__device__ __forceinline__ float gelu_fast(float x) {
    // 0.5x(1+tanh(0.79788456(x+0.044715x^3))) == x*sigmoid(1.59576912x(1+0.044715x^2))
    float u = x * (1.5957691216057308f + 0.0713548135f * x * x);
    float e = __builtin_amdgcn_exp2f(-LOG2E_ * u);
    return x * __builtin_amdgcn_rcpf(1.0f + e);
}

// ---------------------------------------------------------------------------
// CDNA5 async global->LDS copy (16B per lane, no VGPR round-trip, ASYNCcnt)
// ---------------------------------------------------------------------------
__device__ __forceinline__ void async_copy_b128(void* lds_dst, const void* gsrc) {
    unsigned loff = (unsigned)(uintptr_t)(__attribute__((address_space(3))) void*)lds_dst;
    asm volatile("global_load_async_to_lds_b128 %0, %1, off"
                 :: "v"(loff), "v"(gsrc) : "memory");
}
__device__ __forceinline__ void wait_async0() {
    asm volatile("s_wait_asynccnt 0" ::: "memory");
}

// ---------------------------------------------------------------------------
// Workspace layout (float offsets)
// ---------------------------------------------------------------------------
#define WS_XG    ((size_t)0)                    // B*TD*G           = 2097152 f
#define WS_WIHT  ((size_t)2097152)              // 512*128          =   65536 f
#define WS_WHHT  ((size_t)2162688)              // 512*128          =   65536 f
#define WS_DEC   ((size_t)2228224)              // B*TD*H           =  524288 f
#define WS_ENCP  ((size_t)2752512)              // B*TE*H           =  524288 f
#define WS_DECP  ((size_t)3276800)              // B*TD*H           =  524288 f
#define WS_W1SW  ((size_t)3801088)              // 16384 bf16 (as 8192 f)
#define WS_W2SW  ((size_t)3809280)              // 14336 bf16 (as 7168 f)

// ---------------------------------------------------------------------------
// Kernel 0: weight transposes + bf16 WMMA B-tile swizzles
//   B-tile register image: lane holds n = nb*16 + (lane&15),
//   K = kb*32 + (lane>>4)*8 + (t<8 ? t : t+8)  for t=0..15  (32B per lane).
// ---------------------------------------------------------------------------
__global__ void k_prep(const float* __restrict__ Wih, const float* __restrict__ Whh,
                       const float* __restrict__ W1,  const float* __restrict__ W2,
                       float* __restrict__ WihT, float* __restrict__ WhhT,
                       __bf16* __restrict__ w1sw, __bf16* __restrict__ w2sw) {
    int i = blockIdx.x * 256 + threadIdx.x;
    if (i < 65536) {                       // WihT[d*512+g] = Wih[g*128+d]
        int g = i & 511, d = i >> 9;
        WihT[i] = Wih[g * D_ + d];
    } else if (i < 131072) {
        int j = i - 65536;
        int g = j & 511, d = j >> 9;
        WhhT[j] = Whh[g * H_ + d];
    } else if (i < 131072 + 16384) {       // W1 swizzle: j = ((kb*8+nb)*32+lane)*16+t
        int j = i - 131072;
        int t = j & 15, lane = (j >> 4) & 31, nb = (j >> 9) & 7, kb = j >> 12;
        int k = kb * 32 + ((lane >> 4) << 3) + (t < 8 ? t : t + 8);
        int n = nb * 16 + (lane & 15);
        w1sw[j] = (__bf16)W1[k * H_ + n];
    } else if (i < 131072 + 16384 + 14336) { // W2 swizzle (pad n>=100 with 0)
        int j = i - 147456;
        int t = j & 15, lane = (j >> 4) & 31;
        int rem = j >> 9;                   // kb*7 + nb
        int nb = rem % 7, kb = rem / 7;
        int k = kb * 32 + ((lane >> 4) << 3) + (t < 8 ? t : t + 8);
        int n = nb * 16 + (lane & 15);
        w2sw[j] = (n < VP_) ? (__bf16)W2[k * VP_ + n] : (__bf16)0.0f;
    }
}

// ---------------------------------------------------------------------------
// Kernel 1: Xg[b,t,g] = emb[tokens[b,t]] . W_ih[g,:] + (b_ih+b_hh)[g]
// ---------------------------------------------------------------------------
__global__ void k_xgates(const int* __restrict__ tokens, const float* __restrict__ emb,
                         const float* __restrict__ bih, const float* __restrict__ bhh,
                         const float* __restrict__ WihT, float* __restrict__ Xg) {
    __shared__ float xs[D_];
    int blk = blockIdx.x;          // b*TD + t
    int tid = threadIdx.x;         // 256
    if (tid < D_) {
        int tok = tokens[blk];
        xs[tid] = emb[tok * D_ + tid];
    }
    __syncthreads();
    for (int g = tid; g < G_; g += 256) {
        float acc = bih[g] + bhh[g];
        #pragma unroll 8
        for (int d = 0; d < D_; ++d) acc = fmaf(xs[d], WihT[d * G_ + g], acc);
        Xg[(size_t)blk * G_ + g] = acc;
    }
}

// ---------------------------------------------------------------------------
// Kernel 2: sequential LSTM; one block per batch row; W_hh^T resident in
// 256KB of the WGP's 320KB LDS (filled via async global->LDS DMA).
// 128 threads, thread j owns gates 4j..4j+3.
// ---------------------------------------------------------------------------
__global__ void k_lstm(const float* __restrict__ Xg, const int* __restrict__ lengths,
                       const float* __restrict__ h0, const float* __restrict__ c0,
                       const float* __restrict__ WhhT, float* __restrict__ dec) {
    extern __shared__ __align__(16) char dynsm[];
    float* whh = (float*)dynsm;          // 65536 floats = 256KB
    float* hs  = whh + 65536;            // 128
    float* cs  = hs + 128;               // 128
    float* gs  = cs + 128;               // 512
    int b = blockIdx.x, tid = threadIdx.x; // 128 threads

    for (int i = tid; i < 16384; i += 128)
        async_copy_b128(&((float4*)whh)[i], &((const float4*)WhhT)[i]);
    hs[tid] = h0[tid];
    cs[tid] = c0[tid];
    int len = lengths[b]; if (len < 1) len = 1;
    wait_async0();
    __syncthreads();

    for (int t = 0; t < TD_; ++t) {
        const float* xg = Xg + ((size_t)(b * TD_ + t)) * G_ + 4 * tid;
        float4 acc = *(const float4*)xg;
        #pragma unroll 4
        for (int d = 0; d < H_; ++d) {
            float hv = hs[d];
            float4 w = *(const float4*)&whh[d * G_ + 4 * tid];
            acc.x = fmaf(hv, w.x, acc.x);
            acc.y = fmaf(hv, w.y, acc.y);
            acc.z = fmaf(hv, w.z, acc.z);
            acc.w = fmaf(hv, w.w, acc.w);
        }
        *(float4*)&gs[4 * tid] = acc;
        __syncthreads();
        float ig = fast_sigmoid(gs[tid]);
        float fg = fast_sigmoid(gs[H_ + tid]);
        float gg = fast_tanh(gs[2 * H_ + tid]);
        float og = fast_sigmoid(gs[3 * H_ + tid]);
        float cn = fg * cs[tid] + ig * gg;
        float hn = og * fast_tanh(cn);
        bool msk = (t < len);
        hs[tid] = msk ? hn : hs[tid];
        cs[tid] = msk ? cn : cs[tid];
        dec[((size_t)(b * TD_ + t)) * H_ + tid] = msk ? hn : 0.0f;
        __syncthreads();
    }
}

// ---------------------------------------------------------------------------
// Kernel 3: enc_proj = enc @ Wp[:E];  dec_proj = dec @ Wp[E:] + bp
// ---------------------------------------------------------------------------
__global__ void k_proj(const float* __restrict__ enc, const float* __restrict__ dec,
                       const float* __restrict__ Wp, const float* __restrict__ bp,
                       float* __restrict__ encp, float* __restrict__ decp) {
    int blk = blockIdx.x, tid = threadIdx.x; // 128
    if (blk < B_ * TE_) {
        const float* row = enc + (size_t)blk * E_;
        float acc = 0.0f;
        #pragma unroll 8
        for (int e = 0; e < E_; ++e) acc = fmaf(row[e], Wp[e * H_ + tid], acc);
        encp[(size_t)blk * H_ + tid] = acc;
    } else {
        int r = blk - B_ * TE_;
        const float* row = dec + (size_t)r * H_;
        float acc = bp[tid];
        #pragma unroll 8
        for (int k = 0; k < H_; ++k) acc = fmaf(row[k], Wp[(E_ + k) * H_ + tid], acc);
        decp[(size_t)r * H_ + tid] = acc;
    }
}

// ---------------------------------------------------------------------------
// Kernel 4: fused  z = gelu(dec_proj + enc_proj); h1 = gelu(z@W1+b1);
//                  out = (h1@W2+b2) transposed to (B,TE,TD,VP)
// One block per (b,td); 8 waves x 16 te rows; z & h1 never touch HBM.
// Bulk LDS staging via GLOBAL_LOAD_ASYNC_TO_LDS_B128 (ASYNCcnt).
// ---------------------------------------------------------------------------
__global__ void k_fused(const float* __restrict__ encp, const float* __restrict__ decp_g,
                        const __bf16* __restrict__ w1sw, const __bf16* __restrict__ w2sw,
                        const float* __restrict__ b1, const float* __restrict__ b2,
                        float* __restrict__ out) {
    extern __shared__ __align__(16) char dynsm[];
    float*  enc_s = (float*)dynsm;          // 16384 f (64KB)
    float*  dec_s = enc_s + 16384;          // 128
    float*  b1_s  = dec_s + 128;            // 128
    float*  b2_s  = b1_s + 128;             // 128 (padded)
    __bf16* w1_s  = (__bf16*)(b2_s + 128);  // 16384 bf16 (32KB)
    __bf16* w2_s  = w1_s + 16384;           // 14336 bf16 (28KB)
    __bf16* h1_s  = w2_s + 14336;           // 8 waves * 16*128 bf16 (32KB)

    int blk = blockIdx.x;
    int b = blk >> 7, td = blk & 127;
    int tid = threadIdx.x;                  // 256 threads = 8 waves

    // Async DMA: enc_proj tile (64KB) + swizzled bf16 weights (60KB) -> LDS
    const float4* encg = (const float4*)(encp + (size_t)b * TE_ * H_);
    for (int i = tid; i < 4096; i += 256)
        async_copy_b128(&((float4*)enc_s)[i], &encg[i]);
    for (int i = tid; i < 2048; i += 256)
        async_copy_b128((char*)w1_s + i * 16, (const char*)w1sw + i * 16);
    for (int i = tid; i < 1792; i += 256)
        async_copy_b128((char*)w2_s + i * 16, (const char*)w2sw + i * 16);
    if (tid < 128) {
        dec_s[tid] = decp_g[(size_t)(b * TD_ + td) * H_ + tid];
        b1_s[tid]  = b1[tid];
        b2_s[tid]  = (tid < VP_) ? b2[tid] : 0.0f;
    }
    wait_async0();
    __syncthreads();

    int wave = tid >> 5, lane = tid & 31;
    int m = lane & 15, half = lane >> 4;    // A-row / D-column index; lane half
    int te0 = wave * 16;
    const float* erow = enc_s + (te0 + m) * H_;

    // Build z A-tiles in the 16-bit 16x32 A layout:
    // lane covers K = koff..koff+7 (elems 0..7) and koff+16..koff+23 (elems 8..15)
    v16bf az[4];
    #pragma unroll
    for (int kb = 0; kb < 4; ++kb) {
        int koff = kb * 32 + half * 8;
        #pragma unroll
        for (int t = 0; t < 16; ++t) {
            int k = koff + (t < 8 ? t : t + 8);
            float x = dec_s[k] + erow[k];
            az[kb][t] = (__bf16)gelu_fast(x);
        }
    }

    // h1 = gelu(z @ W1 + b1), staged to per-wave LDS in [m][k] order (bf16).
    // Preload all 4 K-tiles per N-tile so the 4 WMMAs issue back-to-back.
    __bf16* h1w = h1_s + wave * 2048;
    #pragma unroll
    for (int nb = 0; nb < 8; ++nb) {
        const __bf16* base = w1_s + ((nb * 32 + lane) << 4);
        v16bf bw0 = *(const v16bf*)(base);
        v16bf bw1 = *(const v16bf*)(base + 4096);   // kb stride = 8*32*16
        v16bf bw2 = *(const v16bf*)(base + 8192);
        v16bf bw3 = *(const v16bf*)(base + 12288);
        v8f acc = {0.f, 0.f, 0.f, 0.f, 0.f, 0.f, 0.f, 0.f};
        acc = __builtin_amdgcn_wmma_f32_16x16x32_bf16(false, az[0], false, bw0, (short)0, acc, false, false);
        acc = __builtin_amdgcn_wmma_f32_16x16x32_bf16(false, az[1], false, bw1, (short)0, acc, false, false);
        acc = __builtin_amdgcn_wmma_f32_16x16x32_bf16(false, az[2], false, bw2, (short)0, acc, false, false);
        acc = __builtin_amdgcn_wmma_f32_16x16x32_bf16(false, az[3], false, bw3, (short)0, acc, false, false);
        int n = nb * 16 + m;                 // D layout: lane's column
        float bias = b1_s[n];
        #pragma unroll
        for (int r = 0; r < 8; ++r) {
            int mm = r + half * 8;           // D layout: row per acc register
            h1w[mm * H_ + n] = (__bf16)gelu_fast(acc[r] + bias);
        }
    }
    // Per-wave LDS handoff (disjoint regions per wave): CDNA5 split DS counter.
    asm volatile("s_wait_dscnt 0" ::: "memory");

    // Reload h1 tile as A-matrices (two contiguous 16B LDS reads per lane/kb)
    v16bf a2[4];
    #pragma unroll
    for (int kb = 0; kb < 4; ++kb) {
        int koff = kb * 32 + half * 8;
        v8bf lo = *(const v8bf*)(h1w + m * H_ + koff);
        v8bf hi = *(const v8bf*)(h1w + m * H_ + koff + 16);
        #pragma unroll
        for (int t = 0; t < 8; ++t) { a2[kb][t] = lo[t]; a2[kb][8 + t] = hi[t]; }
    }

    // scores = h1 @ W2 + b2, written transposed: out[b, te, td, vp]
    #pragma unroll
    for (int nb = 0; nb < 7; ++nb) {
        const __bf16* base = w2_s + ((nb * 32 + lane) << 4);
        v16bf bw0 = *(const v16bf*)(base);
        v16bf bw1 = *(const v16bf*)(base + 3584);   // kb stride = 7*32*16
        v16bf bw2 = *(const v16bf*)(base + 7168);
        v16bf bw3 = *(const v16bf*)(base + 10752);
        v8f acc = {0.f, 0.f, 0.f, 0.f, 0.f, 0.f, 0.f, 0.f};
        acc = __builtin_amdgcn_wmma_f32_16x16x32_bf16(false, a2[0], false, bw0, (short)0, acc, false, false);
        acc = __builtin_amdgcn_wmma_f32_16x16x32_bf16(false, a2[1], false, bw1, (short)0, acc, false, false);
        acc = __builtin_amdgcn_wmma_f32_16x16x32_bf16(false, a2[2], false, bw2, (short)0, acc, false, false);
        acc = __builtin_amdgcn_wmma_f32_16x16x32_bf16(false, a2[3], false, bw3, (short)0, acc, false, false);
        int n = nb * 16 + m;
        if (n < VP_) {
            float bias = b2_s[n];
            #pragma unroll
            for (int r = 0; r < 8; ++r) {
                int te = te0 + r + half * 8;
                out[(((size_t)(b * TE_ + te)) * TD_ + td) * VP_ + n] = acc[r] + bias;
            }
        }
    }
}

// ---------------------------------------------------------------------------
extern "C" void kernel_launch(void* const* d_in, const int* in_sizes, int n_in,
                              void* d_out, int out_size, void* d_ws, size_t ws_size,
                              hipStream_t stream) {
    const int*   tokens  = (const int*)d_in[0];
    const int*   lengths = (const int*)d_in[1];
    const float* enc     = (const float*)d_in[2];
    const float* emb     = (const float*)d_in[3];
    const float* W_ih    = (const float*)d_in[4];
    const float* W_hh    = (const float*)d_in[5];
    const float* b_ih    = (const float*)d_in[6];
    const float* b_hh    = (const float*)d_in[7];
    const float* h0      = (const float*)d_in[8];
    const float* c0      = (const float*)d_in[9];
    const float* Wp      = (const float*)d_in[10];
    const float* bp      = (const float*)d_in[11];
    const float* W1      = (const float*)d_in[12];
    const float* b1      = (const float*)d_in[13];
    const float* W2      = (const float*)d_in[14];
    const float* b2      = (const float*)d_in[15];
    float* out = (float*)d_out;

    float* ws = (float*)d_ws;
    float*  Xg   = ws + WS_XG;
    float*  WihT = ws + WS_WIHT;
    float*  WhhT = ws + WS_WHHT;
    float*  dec  = ws + WS_DEC;
    float*  encp = ws + WS_ENCP;
    float*  decp = ws + WS_DECP;
    __bf16* w1sw = (__bf16*)(ws + WS_W1SW);
    __bf16* w2sw = (__bf16*)(ws + WS_W2SW);

    // 0) transposes + WMMA swizzles  (161792 work items)
    k_prep<<<632, 256, 0, stream>>>(W_ih, W_hh, W1, W2, WihT, WhhT, w1sw, w2sw);
    // 1) input-gate precompute for all timesteps
    k_xgates<<<B_ * TD_, 256, 0, stream>>>(tokens, emb, b_ih, b_hh, WihT, Xg);
    // 2) sequential LSTM, W_hh in 256KB LDS (CDNA5 WGP has 320KB)
    k_lstm<<<B_, 128, (65536 + 128 + 128 + 512) * sizeof(float), stream>>>(
        Xg, lengths, h0, c0, WhhT, dec);
    // 3) encoder/decoder projections through Wp
    k_proj<<<2 * B_ * TE_, 128, 0, stream>>>(enc, dec, Wp, bp, encp, decp);
    // 4) fused pairwise MLP with bf16 WMMA, ~157KB LDS per block
    size_t fused_smem = 16384 * 4 + 3 * 128 * 4 + (16384 + 14336 + 16384) * 2;
    k_fused<<<B_ * TD_, 256, fused_smem, stream>>>(encp, decp, w1sw, w2sw, b1, b2, out);
}
